// GCNRegression_15479062135041
// MI455X (gfx1250) — matrix operand order repaired and checked by reading, not compiled, not run on hardware
//
#include <hip/hip_runtime.h>

typedef __attribute__((ext_vector_type(16))) _Float16 v16h;
typedef __attribute__((ext_vector_type(8)))  _Float16 v8h;
typedef __attribute__((ext_vector_type(4)))  _Float16 v4h;
typedef __attribute__((ext_vector_type(4)))  float    v4f;
typedef __attribute__((ext_vector_type(8)))  float    v8f;

#define N_NODES 50000
#define N_EDGES 1600000
#define C       128
#define G       128

// ---------------------------------------------------------------- utilities
__global__ void zero_f32(float* __restrict__ p, long long n) {
  long long i = (long long)blockIdx.x * blockDim.x + threadIdx.x;
  if (i < n) p[i] = 0.0f;
}

__global__ void edge_degree(const long long* __restrict__ ei, float* __restrict__ deg) {
  int e = blockIdx.x * blockDim.x + threadIdx.x;
  if (e < N_EDGES) atomicAdd(&deg[ei[(long long)N_EDGES + e]], 1.0f);
}

__global__ void make_dinv(float* __restrict__ degdinv) {
  int n = blockIdx.x * blockDim.x + threadIdx.x;
  if (n < N_NODES) degdinv[n] = rsqrtf(degdinv[n] + 1.0f);
}

// Convert W (f32, k-major [k][n]) into fragment-ready f16 layout:
//   Bswz[(((wv*4)+kt)*32 + lane)*16 + h]  with
//   col = wv*16 + (lane&15), hi = lane>>4, k = h + hi*16 + kt*32
// so each wave's B fragment for a given kt is a contiguous, lane-striped 1KB
// chunk -> two global_load_b128 per lane, WGP$-resident.
__global__ void cvt_w_swz(const float* __restrict__ W, _Float16* __restrict__ Bswz) {
  int i = blockIdx.x * blockDim.x + threadIdx.x;   // 0 .. 16383
  if (i >= C * C) return;
  int h    = i & 15;
  int lane = (i >> 4) & 31;
  int kt   = (i >> 9) & 3;
  int wv   = i >> 11;
  int col  = wv * 16 + (lane & 15);
  int hi   = lane >> 4;
  int k    = h + hi * 16 + kt * 32;
  Bswz[i] = (_Float16)W[k * C + col];
}

// ---------------------------------------------------------------- WMMA GEMM
// H[N,128] = A[N,128] @ W[128,128].
// 256 threads = 8 waves; block owns 16 output rows, wave wv owns cols [16wv,16wv+16).
// A: f32 global -> f16 LDS stripe (4KB). B: pre-swizzled f16 fragments in global.
__global__ __launch_bounds__(256)
void gemm_wmma(const float* __restrict__ A, const _Float16* __restrict__ Bswz,
               float* __restrict__ H) {
  __shared__ _Float16 sA[16 * C];   // 4 KB

  const int t    = threadIdx.x;
  const int lane = t & 31;
  const int wv   = t >> 5;
  const long long rowBase = (long long)blockIdx.x * 16;

  // Stage A stripe: 2048 f32 -> f16. Each thread: 2x global_load_b128 +
  // convert + 2x ds_store_b64.
  {
    const v4f* src = (const v4f*)(A + rowBase * C);
    #pragma unroll
    for (int i = 0; i < 2; ++i) {
      int idx = t + i * 256;                 // float4 index, 512 total
      v4f f = src[idx];
      v4h hlf = __builtin_convertvector(f, v4h);
      *(v4h*)(sA + idx * 4) = hlf;
    }
  }
  __syncthreads();

  const int m   = lane & 15;
  const int hi  = lane >> 4;
  const int col = wv * 16 + (lane & 15);

  v8f acc = {};
  #pragma unroll
  for (int kt = 0; kt < 4; ++kt) {          // K = 128 in steps of 32
    // A fragment: two contiguous 8-half groups -> 2x ds_load_b128.
    const _Float16* ap = sA + m * C + hi * 16 + kt * 32;
    v8h alo = *(const v8h*)(ap);
    v8h ahi = *(const v8h*)(ap + 8);
    v16h a = __builtin_shufflevector(alo, ahi,
        0, 1, 2, 3, 4, 5, 6, 7, 8, 9, 10, 11, 12, 13, 14, 15);

    // B fragment: contiguous 32B per lane -> 2x global_load_b128.
    const _Float16* bp = Bswz + (((wv * 4 + kt) * 32 + lane) << 4);
    v8h blo = *(const v8h*)(bp);
    v8h bhi = *(const v8h*)(bp + 8);
    v16h b = __builtin_shufflevector(blo, bhi,
        0, 1, 2, 3, 4, 5, 6, 7, 8, 9, 10, 11, 12, 13, 14, 15);

    acc = __builtin_amdgcn_wmma_f32_16x16x32_f16(
        /*neg_a=*/false, a, /*neg_b=*/false, b,
        /*c_mod=*/(short)0, acc, /*reuse_a=*/false, /*reuse_b=*/false);
  }

  // D layout: VGPR r -> row r + hi*8, col within wave tile.
  #pragma unroll
  for (int r = 0; r < 8; ++r) {
    long long row = rowBase + r + hi * 8;
    H[row * C + col] = acc[r];
  }
}

// ---------------------------------------------------------------- edge scatter
// One wave per edge: lane loads float4 of h[src] (channels 4*lane..4*lane+3),
// scales by dinv[src]*dinv[dst], atomically accumulates into agg[dst] (L2-resident).
__global__ __launch_bounds__(256)
void edge_agg(const float* __restrict__ h, const long long* __restrict__ ei,
              const float* __restrict__ dinv, float* __restrict__ agg) {
  long long gid = (long long)blockIdx.x * blockDim.x + threadIdx.x;
  int e    = (int)(gid >> 5);
  int lane = (int)(gid & 31);
  if (e >= N_EDGES) return;
  long long s = ei[e];
  long long d = ei[(long long)N_EDGES + e];
  float coef = dinv[s] * dinv[d];
  float4 v = ((const float4*)(h + s * C))[lane];
  float* base = agg + d * C + (long long)lane * 4;
  atomicAdd(base + 0, v.x * coef);
  atomicAdd(base + 1, v.y * coef);
  atomicAdd(base + 2, v.z * coef);
  atomicAdd(base + 3, v.w * coef);
}

// out = relu(agg + h*dinv^2 + bias)
__global__ void combine_relu(const float* __restrict__ agg, const float* __restrict__ h,
                             const float* __restrict__ dinv, const float* __restrict__ bias,
                             float* __restrict__ out) {
  long long i = (long long)blockIdx.x * blockDim.x + threadIdx.x;
  if (i >= (long long)N_NODES * C) return;
  int n = (int)(i >> 7);
  float di = dinv[n];
  float v = agg[i] + h[i] * di * di + bias[(int)(i & (C - 1))];
  out[i] = fmaxf(v, 0.0f);
}

// global mean pool (scatter phase)
__global__ void pool_scatter(const float* __restrict__ act, const long long* __restrict__ batch,
                             float* __restrict__ sums, float* __restrict__ counts) {
  long long i = (long long)blockIdx.x * blockDim.x + threadIdx.x;
  if (i >= (long long)N_NODES * C) return;
  int n = (int)(i >> 7);
  int c = (int)(i & (C - 1));
  int g = (int)batch[n];
  atomicAdd(&sums[(long long)g * C + c], act[i]);
  if (c == 0) atomicAdd(&counts[g], 1.0f);
}

// pooled @ Wfc + bfc  (128 graphs x (128 -> 1)); tiny, one block.
__global__ void final_fc(const float* __restrict__ sums, const float* __restrict__ counts,
                         const float* __restrict__ Wfc, const float* __restrict__ bfc,
                         float* __restrict__ out) {
  int g = threadIdx.x;
  if (g >= G) return;
  float cnt = fmaxf(counts[g], 1.0f);
  float acc = bfc[0];
  #pragma unroll 4
  for (int c = 0; c < C; ++c) acc += (sums[g * C + c] / cnt) * Wfc[c];
  out[g] = acc;
}

// ---------------------------------------------------------------- launch
extern "C" void kernel_launch(void* const* d_in, const int* in_sizes, int n_in,
                              void* d_out, int out_size, void* d_ws, size_t ws_size,
                              hipStream_t stream) {
  (void)in_sizes; (void)n_in; (void)out_size; (void)ws_size;
  const float*     x     = (const float*)d_in[0];
  const long long* ei    = (const long long*)d_in[1];   // int64 [2,E]
  const long long* batch = (const long long*)d_in[2];   // int64 [N]
  const float*     W1    = (const float*)d_in[3];
  const float*     b1    = (const float*)d_in[4];
  const float*     W2    = (const float*)d_in[5];
  const float*     b2    = (const float*)d_in[6];
  const float*     Wfc   = (const float*)d_in[7];
  const float*     bfc   = (const float*)d_in[8];
  float*           out   = (float*)d_out;

  // Workspace carve-up (256B-aligned regions).
  char* ws = (char*)d_ws;
  size_t off = 0;
  auto carve = [&](size_t bytes) -> char* {
    char* p = ws + off;
    off += (bytes + 255) & ~(size_t)255;
    return p;
  };
  float*     dinv   = (float*)carve((size_t)N_NODES * 4);
  float*     bufH   = (float*)carve((size_t)N_NODES * C * 4);
  float*     bufAgg = (float*)carve((size_t)N_NODES * C * 4);
  float*     bufAct = (float*)carve((size_t)N_NODES * C * 4);
  _Float16*  W1s    = (_Float16*)carve((size_t)C * C * 2);
  _Float16*  W2s    = (_Float16*)carve((size_t)C * C * 2);
  float*     sums   = (float*)carve((size_t)G * C * 4);
  float*     counts = (float*)carve((size_t)G * 4);

  const long long NC  = (long long)N_NODES * C;
  const int gridN     = (N_NODES + 255) / 256;
  const int gridE     = (N_EDGES + 255) / 256;
  const int gridNC    = (int)((NC + 255) / 256);
  const int gridE32   = (int)(((long long)N_EDGES * 32 + 255) / 256);
  const int gridGemm  = N_NODES / 16;                    // 3125, exact

  // Degrees -> dinv (deg and dinv share a buffer).
  zero_f32<<<gridN, 256, 0, stream>>>(dinv, N_NODES);
  edge_degree<<<gridE, 256, 0, stream>>>(ei, dinv);
  make_dinv<<<gridN, 256, 0, stream>>>(dinv);

  // Pre-swizzle weights into WMMA B-fragment layout (f16).
  cvt_w_swz<<<(C * C + 255) / 256, 256, 0, stream>>>(W1, W1s);
  cvt_w_swz<<<(C * C + 255) / 256, 256, 0, stream>>>(W2, W2s);

  // Layer 1.
  gemm_wmma<<<gridGemm, 256, 0, stream>>>(x, W1s, bufH);
  zero_f32<<<gridNC, 256, 0, stream>>>(bufAgg, NC);
  edge_agg<<<gridE32, 256, 0, stream>>>(bufH, ei, dinv, bufAgg);
  combine_relu<<<gridNC, 256, 0, stream>>>(bufAgg, bufH, dinv, b1, bufAct);

  // Layers 2..4 (conv2 applied NUM_LAYERS=3 times).
  for (int l = 0; l < 3; ++l) {
    gemm_wmma<<<gridGemm, 256, 0, stream>>>(bufAct, W2s, bufH);
    zero_f32<<<gridNC, 256, 0, stream>>>(bufAgg, NC);
    edge_agg<<<gridE32, 256, 0, stream>>>(bufH, ei, dinv, bufAgg);
    combine_relu<<<gridNC, 256, 0, stream>>>(bufAgg, bufH, dinv, b2, bufAct);
  }

  // Global mean pool + FC head.
  zero_f32<<<(G * C + 255) / 256, 256, 0, stream>>>(sums, (long long)G * C);
  zero_f32<<<1, 256, 0, stream>>>(counts, G);
  pool_scatter<<<gridNC, 256, 0, stream>>>(bufAct, batch, sums, counts);
  final_fc<<<1, 128, 0, stream>>>(sums, counts, Wfc, bfc, out);
}
